// NaiveLSTM_75771813036713
// MI455X (gfx1250) — matrix operand order repaired and checked by reading, not compiled, not run on hardware
//
#include <hip/hip_runtime.h>
#include <hip/hip_bf16.h>

// LSTM cell forward for MI455X (gfx1250), wave32 + WMMA, split-bf16 precision.
//
// Fast path (requires ~201MB workspace):
//   1) Pre-convert weights -> per-gate [2048][4096] bf16 hi/lo planes ([Win|Wh]).
//   2) Pre-convert activations -> [4096][4096] bf16 hi/lo planes ([X^T | H]),
//      K-contiguous so the GEMM streams B without transposes.
//   3) Fused GEMM+gating kernel: all staging is b128 (no per-iter conversion
//      VALU), 24 v_wmma_f32_16x16x32_bf16 per wave per K-step.
// Fallback path (small ws): previous fused kernel with in-loop splits.

typedef __attribute__((ext_vector_type(16))) __bf16 v16bf;
typedef __attribute__((ext_vector_type(8)))  __bf16 v8bf;
typedef __attribute__((ext_vector_type(8)))  float  v8f;

#define STRA 40   // LDS row stride in bf16 elements (80B: 16B-aligned, bank-spread)

__device__ __forceinline__ float fsig(float x) {
    return 1.0f / (1.0f + __expf(-x));
}
__device__ __forceinline__ float ftanh_fast(float x) {
    return 2.0f / (1.0f + __expf(-2.0f * x)) - 1.0f;
}

// Split fp32 into truncated-bf16 hi and bf16(lo = x - hi). hi subtraction is exact.
__device__ __forceinline__ void bf_split(float x, unsigned short& hi, unsigned short& lo) {
    unsigned int u = __float_as_uint(x);
    hi = (unsigned short)(u >> 16);
    float r = x - __uint_as_float(u & 0xFFFF0000u);
    lo = (unsigned short)(__float_as_uint(r) >> 16);
}

// Assemble a 16-element bf16 WMMA fragment from two 16-byte LDS chunks.
__device__ __forceinline__ v16bf ld_frag(const unsigned short* p0, const unsigned short* p1) {
    v8bf a = *reinterpret_cast<const v8bf*>(p0);
    v8bf b = *reinterpret_cast<const v8bf*>(p1);
    return __builtin_shufflevector(a, b, 0, 1, 2, 3, 4, 5, 6, 7, 8, 9, 10, 11, 12, 13, 14, 15);
}

// ---------------------------------------------------------------------------
// Pre-pass 1: weights -> wsA planes.
// wsA layout (bf16): plane p of gate g at [(g*2+p)*2048 + m][k], k in [0,4096),
// k<2048 from w_in[g], k>=2048 from w_h[g].
// ---------------------------------------------------------------------------
__global__ __launch_bounds__(256)
void conv_weights_kernel(const float* __restrict__ wi0, const float* __restrict__ wi1,
                         const float* __restrict__ wi2, const float* __restrict__ wi3,
                         const float* __restrict__ wh0, const float* __restrict__ wh1,
                         const float* __restrict__ wh2, const float* __restrict__ wh3,
                         unsigned short* __restrict__ wsA)
{
    const float* Wi[4] = {wi0, wi1, wi2, wi3};
    const float* Wh[4] = {wh0, wh1, wh2, wh3};
    const int k = blockIdx.x * 256 + threadIdx.x;  // 0..4095
    const int m = blockIdx.y;                      // 0..2047
    const int g = blockIdx.z;                      // 0..3
    const float v = (k < 2048) ? Wi[g][(size_t)m * 2048 + k]
                               : Wh[g][(size_t)m * 2048 + (k - 2048)];
    unsigned short h16, l16;
    bf_split(v, h16, l16);
    wsA[((size_t)(g * 2 + 0) * 2048 + m) * 4096 + k] = h16;
    wsA[((size_t)(g * 2 + 1) * 2048 + m) * 4096 + k] = l16;
}

// ---------------------------------------------------------------------------
// Pre-pass 2: H -> wsB columns [2048,4096).  wsB plane p: [p*4096 + n][k].
// ---------------------------------------------------------------------------
__global__ __launch_bounds__(256)
void conv_h_kernel(const float* __restrict__ H, unsigned short* __restrict__ wsB)
{
    const int kk = blockIdx.x * 256 + threadIdx.x; // 0..2047
    const int n  = blockIdx.y;                     // 0..4095
    unsigned short h16, l16;
    bf_split(H[(size_t)n * 2048 + kk], h16, l16);
    wsB[((size_t)(0 * 4096 + n)) * 4096 + 2048 + kk] = h16;
    wsB[((size_t)(1 * 4096 + n)) * 4096 + 2048 + kk] = l16;
}

// ---------------------------------------------------------------------------
// Pre-pass 3: X^T -> wsB columns [0,2048). Tiled 32x32 LDS transpose so both
// global access directions stay coalesced.
// ---------------------------------------------------------------------------
__global__ __launch_bounds__(256)
void conv_x_kernel(const float* __restrict__ X, unsigned short* __restrict__ wsB)
{
    __shared__ float tl[32][33];
    const int tx = threadIdx.x & 31;   // fast index
    const int ty = threadIdx.x >> 5;   // 0..7
    const int n0 = blockIdx.x * 32;    // batch tile
    const int k0 = blockIdx.y * 32;    // input-dim tile
#pragma unroll
    for (int j = 0; j < 4; ++j) {
        const int kk = ty + j * 8;
        tl[kk][tx] = X[(size_t)(k0 + kk) * 4096 + n0 + tx];  // coalesced in n
    }
    __syncthreads();
#pragma unroll
    for (int j = 0; j < 4; ++j) {
        const int nn = ty + j * 8;
        unsigned short h16, l16;
        bf_split(tl[tx][nn], h16, l16);
        wsB[((size_t)(0 * 4096 + n0 + nn)) * 4096 + k0 + tx] = h16; // coalesced in k
        wsB[((size_t)(1 * 4096 + n0 + nn)) * 4096 + k0 + tx] = l16;
    }
}

// ---------------------------------------------------------------------------
// Fast GEMM + LSTM gating. Block tile: 64(hidden) x 64(batch) x all 4 gates.
// 8 waves; wave = (mw, npair): one 16-row M tile x two 16-col N tiles x 4 gates.
// ---------------------------------------------------------------------------
__global__ __launch_bounds__(256)
void lstm_gemm_bf16_kernel(
    const unsigned short* __restrict__ WA,  // [8 planes][2048][4096] bf16
    const unsigned short* __restrict__ WB,  // [2 planes][4096][4096] bf16
    const float* __restrict__ Cin,
    const float* __restrict__ bi0, const float* __restrict__ bh0,
    const float* __restrict__ bi1, const float* __restrict__ bh1,
    const float* __restrict__ bi2, const float* __restrict__ bh2,
    const float* __restrict__ bi3, const float* __restrict__ bh3,
    float* __restrict__ Hout, float* __restrict__ Cout)
{
    __shared__ alignas(16) unsigned short lsA[2][4][64][STRA]; // [plane][gate][m][k]
    __shared__ alignas(16) unsigned short lsB[2][64][STRA];    // [plane][n][k]

    const int t     = threadIdx.x;
    const int lane  = t & 31;
    const int wave  = t >> 5;
    const int mw    = wave & 3;
    const int npair = wave >> 2;
    const int half  = lane >> 4;
    const int lc    = lane & 15;

    const int m0 = blockIdx.y * 64;
    const int n0 = blockIdx.x * 64;

    v8f acc[4][2];
#pragma unroll
    for (int g = 0; g < 4; ++g)
#pragma unroll
        for (int nt = 0; nt < 2; ++nt)
#pragma unroll
            for (int v = 0; v < 8; ++v)
                acc[g][nt][v] = 0.0f;

    // Staging mapping: each thread moves one 16B chunk per (gate,plane) for A
    // (8 chunks) and per plane for B (2 chunks).  r = row in tile, q = k-chunk.
    const int r = t >> 2;  // 0..63
    const int q = t & 3;   // 0..3   (k offset = q*8 bf16 = 16B)

    for (int kt = 0; kt < 128; ++kt) {
        const int kg = kt << 5;

        // ---- global b128 loads into registers (overlap prior WMMAs) ----
        v8bf aS[8], bS[2];
#pragma unroll
        for (int j = 0; j < 8; ++j)
            aS[j] = *reinterpret_cast<const v8bf*>(
                &WA[((size_t)j * 2048 + m0 + r) * 4096 + kg + q * 8]);
#pragma unroll
        for (int j = 0; j < 2; ++j)
            bS[j] = *reinterpret_cast<const v8bf*>(
                &WB[((size_t)j * 4096 + n0 + r) * 4096 + kg + q * 8]);

        if (kt + 1 < 128) {  // prefetch next K tile (global_prefetch_b8)
            __builtin_prefetch(&WA[((size_t)m0 + r) * 4096 + kg + 32 + q * 8], 0, 1);
            __builtin_prefetch(&WB[((size_t)n0 + r) * 4096 + kg + 32 + q * 8], 0, 1);
        }

        __syncthreads();  // previous iteration's fragment reads done

        // ---- b128 stores to LDS ----
#pragma unroll
        for (int j = 0; j < 8; ++j)
            *reinterpret_cast<v8bf*>(&lsA[j & 1][j >> 1][r][q * 8]) = aS[j];
#pragma unroll
        for (int j = 0; j < 2; ++j)
            *reinterpret_cast<v8bf*>(&lsB[j][r][q * 8]) = bS[j];

        __syncthreads();

        // ---- fragments + 24 WMMAs ----
        v16bf bfrag[2][2]; // [ntile][plane]
#pragma unroll
        for (int nt = 0; nt < 2; ++nt) {
            const int n = npair * 32 + nt * 16 + lc;
            bfrag[nt][0] = ld_frag(&lsB[0][n][half * 16], &lsB[0][n][half * 16 + 8]);
            bfrag[nt][1] = ld_frag(&lsB[1][n][half * 16], &lsB[1][n][half * 16 + 8]);
        }
        const int mrow = mw * 16 + lc;
#pragma unroll
        for (int g = 0; g < 4; ++g) {
            v16bf ah = ld_frag(&lsA[0][g][mrow][half * 16], &lsA[0][g][mrow][half * 16 + 8]);
            v16bf al = ld_frag(&lsA[1][g][mrow][half * 16], &lsA[1][g][mrow][half * 16 + 8]);
#pragma unroll
            for (int nt = 0; nt < 2; ++nt) {
                acc[g][nt] = __builtin_amdgcn_wmma_f32_16x16x32_bf16(
                    false, ah, false, bfrag[nt][0], (short)0, acc[g][nt], false, false);
                acc[g][nt] = __builtin_amdgcn_wmma_f32_16x16x32_bf16(
                    false, ah, false, bfrag[nt][1], (short)0, acc[g][nt], false, false);
                acc[g][nt] = __builtin_amdgcn_wmma_f32_16x16x32_bf16(
                    false, al, false, bfrag[nt][0], (short)0, acc[g][nt], false, false);
            }
        }
    }

    // ---- epilogue: each lane owns i,f,g,o for its elements ----
#pragma unroll
    for (int nt = 0; nt < 2; ++nt) {
        const int    n   = n0 + npair * 32 + nt * 16 + lc;
        const size_t row = (size_t)n * 2048;
#pragma unroll
        for (int v = 0; v < 8; ++v) {
            const int m = m0 + mw * 16 + half * 8 + v;
            const float pi = acc[0][nt][v] + bi0[m] + bh0[m];
            const float pf = acc[1][nt][v] + bi1[m] + bh1[m];
            const float pg = acc[2][nt][v] + bi2[m] + bh2[m];
            const float po = acc[3][nt][v] + bi3[m] + bh3[m];
            const float gi = fsig(pi);
            const float gf = fsig(pf);
            const float gg = ftanh_fast(pg);
            const float go = fsig(po);
            const float cold = Cin[row + m];
            const float cnew = gf * cold + gi * gg;
            Cout[row + m] = cnew;
            Hout[row + m] = go * ftanh_fast(cnew);
        }
    }
}

// ---------------------------------------------------------------------------
// Fallback: fully fused kernel (in-loop conversion), used when ws is small.
// ---------------------------------------------------------------------------
__global__ __launch_bounds__(256)
void lstm_cell_wmma_fused_kernel(
    const float* __restrict__ X, const float* __restrict__ H, const float* __restrict__ Cin,
    const float* __restrict__ w0, const float* __restrict__ w1,
    const float* __restrict__ w2, const float* __restrict__ w3,
    const float* __restrict__ w4, const float* __restrict__ w5,
    const float* __restrict__ w6, const float* __restrict__ w7,
    const float* __restrict__ bi0, const float* __restrict__ bh0,
    const float* __restrict__ bi1, const float* __restrict__ bh1,
    const float* __restrict__ bi2, const float* __restrict__ bh2,
    const float* __restrict__ bi3, const float* __restrict__ bh3,
    float* __restrict__ Hout, float* __restrict__ Cout)
{
    __shared__ alignas(16) unsigned short lsA[2][4][64][STRA];
    __shared__ alignas(16) unsigned short lsB[2][64][STRA];

    const int t = threadIdx.x;
    const int lane = t & 31, wave = t >> 5;
    const int mw = wave & 3, npair = wave >> 2;
    const int half = lane >> 4, lc = lane & 15;
    const int m0 = blockIdx.y * 64, n0 = blockIdx.x * 64;

    const float* Wgl[8] = {w0, w1, w2, w3, w4, w5, w6, w7};

    v8f acc[4][2];
#pragma unroll
    for (int g = 0; g < 4; ++g)
#pragma unroll
        for (int nt = 0; nt < 2; ++nt)
#pragma unroll
            for (int v = 0; v < 8; ++v)
                acc[g][nt][v] = 0.0f;

    const int ka = t & 31, ma0 = t >> 5;
    const int nb1 = t & 63, kb1 = t >> 6;
    const int kb2 = t & 31, nb2 = t >> 5;

    for (int kt = 0; kt < 128; ++kt) {
        const int kg = kt << 5;
        const bool phaseIn = (kg < 2048);
        const int kl = phaseIn ? kg : (kg - 2048);

        float aS[4][8], bS[8];
#pragma unroll
        for (int g = 0; g < 4; ++g) {
            const float* W = Wgl[g + (phaseIn ? 0 : 4)];
#pragma unroll
            for (int j = 0; j < 8; ++j)
                aS[g][j] = W[(size_t)(m0 + ma0 + j * 8) * 2048 + (kl + ka)];
        }
        if (phaseIn) {
#pragma unroll
            for (int j = 0; j < 8; ++j)
                bS[j] = X[(size_t)(kg + kb1 + j * 4) * 4096 + (n0 + nb1)];
        } else {
#pragma unroll
            for (int j = 0; j < 8; ++j)
                bS[j] = H[(size_t)(n0 + nb2 + j * 8) * 2048 + (kl + kb2)];
        }

        __syncthreads();
#pragma unroll
        for (int g = 0; g < 4; ++g)
#pragma unroll
            for (int j = 0; j < 8; ++j) {
                unsigned short h16, l16;
                bf_split(aS[g][j], h16, l16);
                lsA[0][g][ma0 + j * 8][ka] = h16;
                lsA[1][g][ma0 + j * 8][ka] = l16;
            }
        if (phaseIn) {
#pragma unroll
            for (int j = 0; j < 8; ++j) {
                unsigned short h16, l16;
                bf_split(bS[j], h16, l16);
                lsB[0][nb1][kb1 + j * 4] = h16;
                lsB[1][nb1][kb1 + j * 4] = l16;
            }
        } else {
#pragma unroll
            for (int j = 0; j < 8; ++j) {
                unsigned short h16, l16;
                bf_split(bS[j], h16, l16);
                lsB[0][nb2 + j * 8][kb2] = h16;
                lsB[1][nb2 + j * 8][kb2] = l16;
            }
        }
        __syncthreads();

        v16bf bfrag[2][2];
#pragma unroll
        for (int nt = 0; nt < 2; ++nt) {
            const int n = npair * 32 + nt * 16 + lc;
            bfrag[nt][0] = ld_frag(&lsB[0][n][half * 16], &lsB[0][n][half * 16 + 8]);
            bfrag[nt][1] = ld_frag(&lsB[1][n][half * 16], &lsB[1][n][half * 16 + 8]);
        }
        const int mrow = mw * 16 + lc;
#pragma unroll
        for (int g = 0; g < 4; ++g) {
            v16bf ah = ld_frag(&lsA[0][g][mrow][half * 16], &lsA[0][g][mrow][half * 16 + 8]);
            v16bf al = ld_frag(&lsA[1][g][mrow][half * 16], &lsA[1][g][mrow][half * 16 + 8]);
#pragma unroll
            for (int nt = 0; nt < 2; ++nt) {
                acc[g][nt] = __builtin_amdgcn_wmma_f32_16x16x32_bf16(
                    false, ah, false, bfrag[nt][0], (short)0, acc[g][nt], false, false);
                acc[g][nt] = __builtin_amdgcn_wmma_f32_16x16x32_bf16(
                    false, ah, false, bfrag[nt][1], (short)0, acc[g][nt], false, false);
                acc[g][nt] = __builtin_amdgcn_wmma_f32_16x16x32_bf16(
                    false, al, false, bfrag[nt][0], (short)0, acc[g][nt], false, false);
            }
        }
    }

#pragma unroll
    for (int nt = 0; nt < 2; ++nt) {
        const int    n   = n0 + npair * 32 + nt * 16 + lc;
        const size_t row = (size_t)n * 2048;
#pragma unroll
        for (int v = 0; v < 8; ++v) {
            const int m = m0 + mw * 16 + half * 8 + v;
            const float pi = acc[0][nt][v] + bi0[m] + bh0[m];
            const float pf = acc[1][nt][v] + bi1[m] + bh1[m];
            const float pg = acc[2][nt][v] + bi2[m] + bh2[m];
            const float po = acc[3][nt][v] + bi3[m] + bh3[m];
            const float gi = fsig(pi), gf = fsig(pf);
            const float gg = ftanh_fast(pg), go = fsig(po);
            const float cold = Cin[row + m];
            const float cnew = gf * cold + gi * gg;
            Cout[row + m] = cnew;
            Hout[row + m] = go * ftanh_fast(cnew);
        }
    }
}

extern "C" void kernel_launch(void* const* d_in, const int* in_sizes, int n_in,
                              void* d_out, int out_size, void* d_ws, size_t ws_size,
                              hipStream_t stream) {
    (void)in_sizes; (void)n_in; (void)out_size;

    const float* X   = (const float*)d_in[0];
    const float* H   = (const float*)d_in[1];
    const float* C   = (const float*)d_in[2];
    const float* wii = (const float*)d_in[3];
    const float* wif = (const float*)d_in[4];
    const float* wig = (const float*)d_in[5];
    const float* wio = (const float*)d_in[6];
    const float* whi = (const float*)d_in[7];
    const float* whf = (const float*)d_in[8];
    const float* whg = (const float*)d_in[9];
    const float* who = (const float*)d_in[10];
    const float* bii = (const float*)d_in[11];
    const float* bhi = (const float*)d_in[12];
    const float* bif = (const float*)d_in[13];
    const float* bhf = (const float*)d_in[14];
    const float* big = (const float*)d_in[15];
    const float* bhg = (const float*)d_in[16];
    const float* bio = (const float*)d_in[17];
    const float* bho = (const float*)d_in[18];

    float* Hout = (float*)d_out;
    float* Cout = (float*)d_out + (size_t)4096 * 2048;

    const size_t elemsA = (size_t)8 * 2048 * 4096;  // 8 weight planes
    const size_t elemsB = (size_t)2 * 4096 * 4096;  // 2 activation planes
    const size_t need   = (elemsA + elemsB) * sizeof(unsigned short); // ~201.3 MB

    if (ws_size >= need) {
        unsigned short* wsA = (unsigned short*)d_ws;
        unsigned short* wsB = wsA + elemsA;

        conv_weights_kernel<<<dim3(16, 2048, 4), 256, 0, stream>>>(
            wii, wif, wig, wio, whi, whf, whg, who, wsA);
        conv_h_kernel<<<dim3(8, 4096), 256, 0, stream>>>(H, wsB);
        conv_x_kernel<<<dim3(128, 64), 256, 0, stream>>>(X, wsB);

        lstm_gemm_bf16_kernel<<<dim3(64, 32), 256, 0, stream>>>(
            wsA, wsB, C,
            bii, bhi, bif, bhf, big, bhg, bio, bho,
            Hout, Cout);
    } else {
        lstm_cell_wmma_fused_kernel<<<dim3(64, 32), 256, 0, stream>>>(
            X, H, C,
            wii, wif, wig, wio, whi, whf, whg, who,
            bii, bhi, bif, bhf, big, bhg, bio, bho,
            Hout, Cout);
    }
}
